// MultiheadAttentionSimple_41283225649362
// MI455X (gfx1250) — matrix operand order repaired and checked by reading, not compiled, not run on hardware
//
#include <hip/hip_runtime.h>
#include <hip/hip_bf16.h>
#include <math.h>

// ---------------------------------------------------------------------------
// Relative-position multihead attention (Transformer-XL style) for gfx1250.
// L=512, B=8, E=1024, H=16, D=64.  All GEMMs via v_wmma_f32_16x16x32_f16.
// ---------------------------------------------------------------------------

typedef __attribute__((ext_vector_type(16))) _Float16 v16h;
typedef __attribute__((ext_vector_type(8)))  float    v8f;

#define L_SEQ 512
#define BSZ   8
#define EDIM  1024
#define NH    16
#define HD    64
#define BH    (BSZ * NH)
#define MROWS (L_SEQ * BSZ)   // 4096 rows for projections

__device__ __forceinline__ v8f zero8() {
  v8f z = {0.f, 0.f, 0.f, 0.f, 0.f, 0.f, 0.f, 0.f};
  return z;
}

__device__ __forceinline__ v8f wmma16(v16h a, v16h b, v8f c) {
  // D = A(16x32 f16) x B(32x16 f16) + C(16x16 f32)
  return __builtin_amdgcn_wmma_f32_16x16x32_f16(
      /*neg_a=*/false, a, /*neg_b=*/false, b,
      /*c_mod=*/(short)0, c, /*reuse_a=*/false, /*reuse_b=*/false);
}

// A fragment (16x32, f16) from row-major [rows][stride] source.
// ISA layout: lane m = lane&15; element e=2r+j -> k = (r>=4?16:0)+half*8+(r&3)*2+j
__device__ __forceinline__ v16h load_fragA(const _Float16* __restrict__ p,
                                           int row0, int stride, int k0,
                                           int lane16, int half_) {
  const _Float16* base = p + (long)(row0 + lane16) * stride + k0;
  v16h a;
#pragma unroll
  for (int r = 0; r < 8; ++r) {
    int k = ((r & 4) << 2) + half_ * 8 + ((r & 3) << 1);
    a[2 * r]     = base[k];
    a[2 * r + 1] = base[k + 1];
  }
  return a;
}

// B fragment (32x16, f16): B[k][n] = src[(nrow0+n)*stride + k0 + k]
// ISA layout: n = lane&15; k = (e>=8?16:0) + (lane>>4)*8 + (e&7)  (e = element 0..15)
__device__ __forceinline__ v16h load_fragB(const _Float16* __restrict__ p,
                                           int nrow0, int stride, int k0,
                                           int lane16, int half_) {
  const _Float16* base = p + (long)(nrow0 + lane16) * stride + k0;
  v16h b;
#pragma unroll
  for (int e = 0; e < 16; ++e) b[e] = base[((e & 8) << 1) + half_ * 8 + (e & 7)];
  return b;
}

// B fragment from pe table [2L][64] with row (=n index, the rel-pos l) clamped.
__device__ __forceinline__ v16h load_fragB_pe(const _Float16* __restrict__ pe,
                                              int l0, int d0,
                                              int lane16, int half_) {
  int l = l0 + lane16;
  l = l < 0 ? 0 : (l > 2 * L_SEQ - 1 ? 2 * L_SEQ - 1 : l);
  const _Float16* base = pe + (long)l * HD + d0;
  v16h b;
#pragma unroll
  for (int e = 0; e < 16; ++e) b[e] = base[((e & 8) << 1) + half_ * 8 + (e & 7)];
  return b;
}

// ---------------------------------------------------------------------------
// Elementwise f32 -> f16 conversion
// ---------------------------------------------------------------------------
__global__ void __launch_bounds__(256) cvt_f32_f16(const float* __restrict__ src,
                                                   _Float16* __restrict__ dst,
                                                   long n) {
  long i = (long)blockIdx.x * 256 + threadIdx.x;
  if (i < n) dst[i] = (_Float16)src[i];
}

// ---------------------------------------------------------------------------
// Sinusoid relative-position table pe[2L][D] (f16), positions -L..L-1
// ---------------------------------------------------------------------------
__global__ void __launch_bounds__(256) pe_kernel(_Float16* __restrict__ pe) {
  int i = blockIdx.x * 256 + threadIdx.x;
  if (i >= 2 * L_SEQ * HD) return;
  int l = i >> 6, d = i & 63;
  int j = d & 31;                                   // half = D/2 = 32
  float inv = expf((float)j * (-logf(10000.0f) / 31.0f));
  float ang = (float)(l - L_SEQ) * inv;
  pe[i] = (_Float16)((d < 32) ? sinf(ang) : cosf(ang));
}

// ---------------------------------------------------------------------------
// Generic projection GEMM: out[m][n] = sum_i A16[m][i] * W16[n][i] (+bias)
// One wave computes a 32x64 tile (2 A-frags reuse each B-frag -> halves
// L2 B traffic; 8 WMMAs per 32-deep K step).  Epilogue by mode:
//  0 = Q : qh = (acc+bq)*scale  -> [bh][l][d];  qbh = qh + r_w_bias
//  1 = K : kh  = acc+bk         -> [bh][l][d]
//  2 = V : vT  = acc+bv         -> [bh][d][l]   (transposed for B-frag loads)
//  3 = O : f32 d_out[m][n] = acc + bo
// ---------------------------------------------------------------------------
__global__ void __launch_bounds__(32) gemm_proj(
    const _Float16* __restrict__ A16, const _Float16* __restrict__ W16,
    const float* __restrict__ bias, const float* __restrict__ rwb,
    _Float16* __restrict__ out_a, _Float16* __restrict__ out_b,
    float* __restrict__ out_f32, int mode) {
  int lane16 = threadIdx.x & 15, half_ = threadIdx.x >> 4;
  int m0 = blockIdx.x * 32;      // row tile (rows are (l*B+b))
  int n0 = blockIdx.y * 64;      // feature tile

  v8f acc[2][4] = {{zero8(), zero8(), zero8(), zero8()},
                   {zero8(), zero8(), zero8(), zero8()}};
  for (int kk = 0; kk < EDIM; kk += 32) {
    v16h a0 = load_fragA(A16, m0,      EDIM, kk, lane16, half_);
    v16h a1 = load_fragA(A16, m0 + 16, EDIM, kk, lane16, half_);
#pragma unroll
    for (int t = 0; t < 4; ++t) {
      v16h b = load_fragB(W16, n0 + t * 16, EDIM, kk, lane16, half_);
      acc[0][t] = wmma16(a0, b, acc[0][t]);
      acc[1][t] = wmma16(a1, b, acc[1][t]);
    }
  }

  const float scaling = 0.125f;  // D^-0.5 = 1/8
#pragma unroll
  for (int t = 0; t < 4; ++t) {
    int ncol = n0 + t * 16 + lane16;       // output feature o
    int h = ncol >> 6, d = ncol & 63;
    float bval = bias[ncol];
#pragma unroll
    for (int sub = 0; sub < 2; ++sub) {
#pragma unroll
      for (int v = 0; v < 8; ++v) {
        int m = m0 + sub * 16 + v + half_ * 8;   // = l*B + b
        int lq = m >> 3, b = m & 7;
        float val = acc[sub][t][v] + bval;
        if (mode == 0) {
          float qs = val * scaling;
          long idx = (((long)b * NH + h) * L_SEQ + lq) * HD + d;
          out_a[idx] = (_Float16)qs;
          out_b[idx] = (_Float16)(qs + rwb[h * HD + d]);
        } else if (mode == 1) {
          long idx = (((long)b * NH + h) * L_SEQ + lq) * HD + d;
          out_a[idx] = (_Float16)val;
        } else if (mode == 2) {
          long idx = (((long)b * NH + h) * HD + d) * L_SEQ + lq;
          out_a[idx] = (_Float16)val;
        } else {
          out_f32[(long)m * EDIM + ncol] = val;
        }
      }
    }
  }
}

// ---------------------------------------------------------------------------
// Fused relative attention, flash-style online softmax.
// Grid (L/16, B*H); one wave per (q-tile, head).
//   S[q,k] = q.k  +  scale*( qb[q].pe[L+k-q] + k[k].pe[L+q-k] )
// Positional terms: 79-wide diagonal band computed with WMMA into LDS,
// gathered diagonally into the score tile.
// ---------------------------------------------------------------------------
__global__ void __launch_bounds__(32) attn_kernel(
    const _Float16* __restrict__ qh, const _Float16* __restrict__ qbh,
    const _Float16* __restrict__ kh, const _Float16* __restrict__ vT,
    const _Float16* __restrict__ pe, const unsigned char* __restrict__ mask,
    _Float16* __restrict__ attn_out) {
  __shared__ float    bf[16][80];      // qb . pe band  (this q-tile)
  __shared__ float    ef[64][80];      // k  . pe band  (this k-tile)
  __shared__ _Float16 pbuf[16][64];    // softmax probs, C-layout -> A-layout

  int lane16 = threadIdx.x & 15, half_ = threadIdx.x >> 4;
  int qbase = blockIdx.x * 16;
  int bh = blockIdx.y, b = bh >> 4, h = bh & 15;
  const float scaling = 0.125f;

  const _Float16* qhh  = qh  + (long)bh * L_SEQ * HD;
  const _Float16* qbhh = qbh + (long)bh * L_SEQ * HD;
  const _Float16* khh  = kh  + (long)bh * L_SEQ * HD;
  const _Float16* vth  = vT  + (long)bh * HD * L_SEQ;
  const unsigned char* mrow = mask + (long)b * L_SEQ;

  // Q fragments (rows qbase..qbase+15, d split 0..31 / 32..63)
  v16h QA0 = load_fragA(qhh,  qbase, HD, 0,  lane16, half_);
  v16h QA1 = load_fragA(qhh,  qbase, HD, 32, lane16, half_);
  v16h QB0 = load_fragA(qbhh, qbase, HD, 0,  lane16, half_);
  v16h QB1 = load_fragA(qbhh, qbase, HD, 32, lane16, half_);

  v8f O[4] = {zero8(), zero8(), zero8(), zero8()};
  float mrun[8], lrun[8];
#pragma unroll
  for (int v = 0; v < 8; ++v) { mrun[v] = -1e30f; lrun[v] = 0.f; }

  for (int kbase = 0; kbase < L_SEQ; kbase += 64) {
    // ---- K A-fragments for this k-tile (also reused by E band) ----------
    v16h ea[4][2];
#pragma unroll
    for (int kt = 0; kt < 4; ++kt) {
      ea[kt][0] = load_fragA(khh, kbase + kt * 16, HD, 0,  lane16, half_);
      ea[kt][1] = load_fragA(khh, kbase + kt * 16, HD, 32, lane16, half_);
    }

    // ---- content scores S = Q @ K^T -------------------------------------
    v8f S[4] = {zero8(), zero8(), zero8(), zero8()};
#pragma unroll
    for (int t = 0; t < 4; ++t) {
      v16h kb0 = load_fragB(khh, kbase + t * 16, HD, 0,  lane16, half_);
      v16h kb1 = load_fragB(khh, kbase + t * 16, HD, 32, lane16, half_);
      S[t] = wmma16(QA0, kb0, S[t]);
      S[t] = wmma16(QA1, kb1, S[t]);
    }

    // ---- B band: bf[qq][c] = qb[qbase+qq] . pe[lb+c],  c = kk+15-qq -----
    int lb = L_SEQ + kbase - qbase - 15;
#pragma unroll
    for (int ct = 0; ct < 5; ++ct) {
      v8f c = zero8();
      v16h p0 = load_fragB_pe(pe, lb + ct * 16, 0,  lane16, half_);
      v16h p1 = load_fragB_pe(pe, lb + ct * 16, 32, lane16, half_);
      c = wmma16(QB0, p0, c);
      c = wmma16(QB1, p1, c);
#pragma unroll
      for (int v = 0; v < 8; ++v)
        bf[v + half_ * 8][ct * 16 + lane16] = c[v];
    }

    // ---- E band: ef[kk][c'] = k[kbase+kk] . pe[lb2+c'], c' = qq+63-kk ---
    // ct-outer so each pe fragment pair is loaded once and reused 4x.
    int lb2 = L_SEQ + qbase - kbase - 63;
#pragma unroll
    for (int ct = 0; ct < 5; ++ct) {
      v16h p0 = load_fragB_pe(pe, lb2 + ct * 16, 0,  lane16, half_);
      v16h p1 = load_fragB_pe(pe, lb2 + ct * 16, 32, lane16, half_);
#pragma unroll
      for (int kt = 0; kt < 4; ++kt) {
        v8f c = zero8();
        c = wmma16(ea[kt][0], p0, c);
        c = wmma16(ea[kt][1], p1, c);
#pragma unroll
        for (int v = 0; v < 8; ++v)
          ef[kt * 16 + v + half_ * 8][ct * 16 + lane16] = c[v];
      }
    }

    // ---- combine + key padding mask -------------------------------------
#pragma unroll
    for (int t = 0; t < 4; ++t) {
#pragma unroll
      for (int v = 0; v < 8; ++v) {
        int qq = v + half_ * 8;
        int kk = t * 16 + lane16;
        float pos = bf[qq][kk + 15 - qq] + ef[kk][qq + 63 - kk];
        float s = S[t][v] + scaling * pos;
        s = mrow[kbase + kk] ? -1e30f : s;
        S[t][v] = s;
      }
    }

    // ---- online softmax --------------------------------------------------
    float mnew[8], alpha[8], psum[8];
#pragma unroll
    for (int v = 0; v < 8; ++v) {
      float mx = fmaxf(fmaxf(S[0][v], S[1][v]), fmaxf(S[2][v], S[3][v]));
#pragma unroll
      for (int off = 1; off < 16; off <<= 1)
        mx = fmaxf(mx, __shfl_xor(mx, off, 32));
      mnew[v] = fmaxf(mrun[v], mx);
      alpha[v] = __expf(mrun[v] - mnew[v]);
      mrun[v] = mnew[v];
      psum[v] = 0.f;
    }
#pragma unroll
    for (int t = 0; t < 4; ++t)
#pragma unroll
      for (int v = 0; v < 8; ++v) {
        float p = __expf(S[t][v] - mnew[v]);
        S[t][v] = p;
        psum[v] += p;
      }
#pragma unroll
    for (int v = 0; v < 8; ++v) {
#pragma unroll
      for (int off = 1; off < 16; off <<= 1)
        psum[v] += __shfl_xor(psum[v], off, 32);
      lrun[v] = lrun[v] * alpha[v] + psum[v];
    }
#pragma unroll
    for (int ot = 0; ot < 4; ++ot)
#pragma unroll
      for (int v = 0; v < 8; ++v) O[ot][v] *= alpha[v];

    // ---- P: C-layout -> LDS -> A-layout ---------------------------------
#pragma unroll
    for (int t = 0; t < 4; ++t)
#pragma unroll
      for (int v = 0; v < 8; ++v)
        pbuf[v + half_ * 8][t * 16 + lane16] = (_Float16)S[t][v];

    v16h PA0, PA1;
#pragma unroll
    for (int r = 0; r < 8; ++r) {
      int k = ((r & 4) << 2) + half_ * 8 + ((r & 3) << 1);
      PA0[2 * r]     = pbuf[lane16][k];
      PA0[2 * r + 1] = pbuf[lane16][k + 1];
      PA1[2 * r]     = pbuf[lane16][32 + k];
      PA1[2 * r + 1] = pbuf[lane16][32 + k + 1];
    }

    // ---- O += P @ V  (V^T layout: rows = d, cols = keys) ----------------
#pragma unroll
    for (int ot = 0; ot < 4; ++ot) {
      v16h vb0 = load_fragB(vth, ot * 16, L_SEQ, kbase,      lane16, half_);
      v16h vb1 = load_fragB(vth, ot * 16, L_SEQ, kbase + 32, lane16, half_);
      O[ot] = wmma16(PA0, vb0, O[ot]);
      O[ot] = wmma16(PA1, vb1, O[ot]);
    }
  }

  // ---- epilogue: normalize, write [ (l*B+b) ][ h*64+d ] as f16 ----------
#pragma unroll
  for (int ot = 0; ot < 4; ++ot) {
#pragma unroll
    for (int v = 0; v < 8; ++v) {
      int qq = v + half_ * 8;
      long lq = qbase + qq;
      float denom = lrun[v] > 0.f ? lrun[v] : 1.0f;
      float o = O[ot][v] / denom;
      long col = (long)h * HD + ot * 16 + lane16;
      attn_out[(lq * BSZ + b) * EDIM + col] = (_Float16)o;
    }
  }
}

// ---------------------------------------------------------------------------
// Host-side orchestration
// ---------------------------------------------------------------------------
extern "C" void kernel_launch(void* const* d_in, const int* in_sizes, int n_in,
                              void* d_out, int out_size, void* d_ws, size_t ws_size,
                              hipStream_t stream) {
  const float* query = (const float*)d_in[0];
  const float* key   = (const float*)d_in[1];
  const float* value = (const float*)d_in[2];
  const unsigned char* mask = (const unsigned char*)d_in[3];
  const float* Wq = (const float*)d_in[4];
  const float* bq = (const float*)d_in[5];
  const float* Wk = (const float*)d_in[6];
  const float* bk = (const float*)d_in[7];
  const float* Wv = (const float*)d_in[8];
  const float* bv = (const float*)d_in[9];
  const float* Wo = (const float*)d_in[10];
  const float* bo = (const float*)d_in[11];
  const float* rwb = (const float*)d_in[12];

  // workspace layout (f16 elements)
  char* ws = (char*)d_ws;
  const long NX = (long)MROWS * EDIM;      // 4096*1024
  const long NW = (long)EDIM * EDIM;       // 1024*1024
  const long NH16 = (long)BH * L_SEQ * HD; // 128*512*64 == NX
  _Float16* xq16 = (_Float16*)ws;
  _Float16* xk16 = xq16 + NX;
  _Float16* xv16 = xk16 + NX;
  _Float16* wq16 = xv16 + NX;
  _Float16* wk16 = wq16 + NW;
  _Float16* wv16 = wk16 + NW;
  _Float16* wo16 = wv16 + NW;
  _Float16* pe16 = wo16 + NW;
  _Float16* qhp  = pe16 + (long)2 * L_SEQ * HD;
  _Float16* qbhp = qhp  + NH16;
  _Float16* khp  = qbhp + NH16;
  _Float16* vtp  = khp  + NH16;
  _Float16* aout = vtp  + NH16;            // ends at ~72 MB

  // 1) convert activations & weights to f16
  cvt_f32_f16<<<(NX + 255) / 256, 256, 0, stream>>>(query, xq16, NX);
  cvt_f32_f16<<<(NX + 255) / 256, 256, 0, stream>>>(key,   xk16, NX);
  cvt_f32_f16<<<(NX + 255) / 256, 256, 0, stream>>>(value, xv16, NX);
  cvt_f32_f16<<<(NW + 255) / 256, 256, 0, stream>>>(Wq, wq16, NW);
  cvt_f32_f16<<<(NW + 255) / 256, 256, 0, stream>>>(Wk, wk16, NW);
  cvt_f32_f16<<<(NW + 255) / 256, 256, 0, stream>>>(Wv, wv16, NW);
  cvt_f32_f16<<<(NW + 255) / 256, 256, 0, stream>>>(Wo, wo16, NW);

  // 2) sinusoid table
  pe_kernel<<<(2 * L_SEQ * HD + 255) / 256, 256, 0, stream>>>(pe16);

  // 3) QKV projections (WMMA)
  dim3 gp(MROWS / 32, EDIM / 64);
  gemm_proj<<<gp, 32, 0, stream>>>(xq16, wq16, bq, rwb, qhp, qbhp, nullptr, 0);
  gemm_proj<<<gp, 32, 0, stream>>>(xk16, wk16, bk, nullptr, khp, nullptr, nullptr, 1);
  gemm_proj<<<gp, 32, 0, stream>>>(xv16, wv16, bv, nullptr, vtp, nullptr, nullptr, 2);

  // 4) fused relative attention (WMMA + flash softmax)
  attn_kernel<<<dim3(L_SEQ / 16, BH), 32, 0, stream>>>(
      qhp, qbhp, khp, vtp, pe16, mask, aout);

  // 5) output projection -> f32 d_out [L,B,E]
  gemm_proj<<<gp, 32, 0, stream>>>(aout, wo16, bo, nullptr, nullptr, nullptr,
                                   (float*)d_out, 3);
}
